// EFFGraphConv_60447369724157
// MI455X (gfx1250) — compile-verified, matrix operand order checked by reading, not run on hardware
//
#include <hip/hip_runtime.h>
#include <hip/hip_bf16.h>

#define DD 64
#define DIN 192
#define NN 30000
#define NE 480000
#define ET 32

typedef __attribute__((ext_vector_type(16))) _Float16 v16h;
typedef __attribute__((ext_vector_type(8)))  float    v8f;

union FragH { v16h h; unsigned int u[8]; };

__device__ __forceinline__ float siluf(float x)  { return x / (1.f + __expf(-x)); }
__device__ __forceinline__ float sigmf(float x)  { return 1.f / (1.f + __expf(-x)); }

__device__ __forceinline__ v8f wmma_f16(v16h a, v16h b, v8f c) {
  return __builtin_amdgcn_wmma_f32_16x16x32_f16(false, a, false, b, (short)0, c,
                                                false, false);
}

// K index inside a 16x32 (f16) A/B fragment: lane half selects K-subrange,
// VGPR v<4 covers K 0..15 half, v>=4 covers K 16..31 half, 2 f16 per VGPR.
__device__ __forceinline__ int kmap(int lane, int j) {
  int kh8 = (lane >> 4) << 3;
  int v = j >> 1, ip = j & 1;
  return (v < 4) ? (kh8 + 2 * v + ip) : (16 + kh8 + 2 * (v - 4) + ip);
}

// ---------------- prep kernels ----------------
__global__ __launch_bounds__(256) void zero_f32(float* p, int n) {
  int i = blockIdx.x * 256 + threadIdx.x;
  if (i < n) p[i] = 0.f;
}

// src: row-major [K][64] f32 weight.  dst: per-fragment swizzled f16 so a lane
// reads its 16 f16 B-operand values as one contiguous 32B chunk.
__global__ __launch_bounds__(256) void swz_weight(const float* __restrict__ src,
                                                  _Float16* __restrict__ dst, int K) {
  int id = blockIdx.x * 256 + threadIdx.x;
  int total = K * DD;                    // (K/32) kslices * 4 ntiles * 32 lanes * 16
  if (id >= total) return;
  int s = id >> 11;                      // kslice
  int t = (id >> 9) & 3;                 // ntile
  int l = (id >> 4) & 31;                // lane
  int j = id & 15;
  int k = s * 32 + kmap(l, j);
  int n = t * 16 + (l & 15);
  dst[id] = (_Float16)src[k * DD + n];
}

// ---------------- fragment loaders ----------------
template <int STRIDE>
__device__ __forceinline__ v16h load_a(const _Float16 (*X)[STRIDE], int mbase,
                                       int lane, int kh8, int kbase) {
  FragH f;
  const int row = mbase + (lane & 15);
#pragma unroll
  for (int v = 0; v < 8; ++v) {
    int kp = (v < 4) ? (kh8 + 2 * v) : (16 + kh8 + 2 * (v - 4));
    f.u[v] = *(const unsigned int*)&X[row][kbase + kp];
  }
  return f.h;
}

__device__ __forceinline__ v16h load_b(const _Float16* __restrict__ w, int s,
                                       int nt, int lane) {
  return *(const v16h*)(w + (size_t)((((s << 2) + nt) << 5) + lane) * 16);
}

// ---------------- fused edge kernel ----------------
__global__ __launch_bounds__(256) void edge_kernel(
    const float* __restrict__ node_feat, const float* __restrict__ edge_feat,
    const int* __restrict__ src, const int* __restrict__ dst,
    const float* __restrict__ eLb0, const float* __restrict__ eLb1,
    const float* __restrict__ eGb0, const float* __restrict__ eGb1,
    const float* __restrict__ nLb0, const float* __restrict__ nLb1,
    const float* __restrict__ nGb0, const float* __restrict__ nGb1,
    const _Float16* __restrict__ wswz, float* __restrict__ e_out,
    float* __restrict__ agg) {
  __shared__ _Float16 xin[ET][DIN];   // [h_src || e || h_dst] as f16
  __shared__ _Float16 hidL[ET][DD];
  __shared__ _Float16 hidG[ET][DD];
  __shared__ int sIdx[ET], dIdx[ET];

  const int tid = threadIdx.x;
  const int e0 = blockIdx.x * ET;
  if (tid < ET) { sIdx[tid] = src[e0 + tid]; dIdx[tid] = dst[e0 + tid]; }
  __syncthreads();

  // gather stage: fp32 global -> f16 LDS
  for (int i = tid; i < ET * DIN; i += 256) {
    int m = i / DIN, c = i % DIN;
    float v;
    if (c < DD)          v = node_feat[(size_t)sIdx[m] * DD + c];
    else if (c < 2 * DD) v = edge_feat[(size_t)(e0 + m) * DD + (c - DD)];
    else                 v = node_feat[(size_t)dIdx[m] * DD + (c - 2 * DD)];
    xin[m][c] = (_Float16)v;
  }
  __syncthreads();

  const int lane = tid & 31;
  const int wave = tid >> 5;
  const int mt = wave >> 2;              // 0..1  -> edge rows [mt*16, mt*16+16)
  const int nt = wave & 3;               // 0..3  -> out cols  [nt*16, nt*16+16)
  const int ncol = nt * 16 + (lane & 15);
  const int kh8 = (lane >> 4) << 3;
  const int mbase = mt * 16;

  const _Float16* eLW0 = wswz;
  const _Float16* eGW0 = wswz + 12288;
  const _Float16* eLW1 = wswz + 24576;
  const _Float16* eGW1 = wswz + 28672;
  const _Float16* nLW0 = wswz + 32768;
  const _Float16* nGW0 = wswz + 45056;
  const _Float16* nLW1 = wswz + 57344;
  const _Float16* nGW1 = wswz + 61440;

  // ===== edge gated-MLP, layer 0 (192 -> 64), both branches =====
  v8f accL = {}, accG = {};
#pragma unroll
  for (int s = 0; s < 6; ++s) {
    v16h a = load_a<DIN>(xin, mbase, lane, kh8, s * 32);
    accL = wmma_f16(a, load_b(eLW0, s, nt, lane), accL);
    accG = wmma_f16(a, load_b(eGW0, s, nt, lane), accG);
  }
  {
    float bL = eLb0[ncol], bG = eGb0[ncol];
#pragma unroll
    for (int r = 0; r < 8; ++r) {
      int ml = mbase + kh8 + r;
      hidL[ml][ncol] = (_Float16)siluf(accL[r] + bL);
      hidG[ml][ncol] = (_Float16)siluf(accG[r] + bG);
    }
  }
  __syncthreads();

  // ===== edge gated-MLP, layer 1 (64 -> 64), residual, e_new =====
  v8f a2L = {}, a2G = {};
#pragma unroll
  for (int s = 0; s < 2; ++s) {
    a2L = wmma_f16(load_a<DD>(hidL, mbase, lane, kh8, s * 32),
                   load_b(eLW1, s, nt, lane), a2L);
    a2G = wmma_f16(load_a<DD>(hidG, mbase, lane, kh8, s * 32),
                   load_b(eGW1, s, nt, lane), a2G);
  }
  {
    float bL = eLb1[ncol], bG = eGb1[ncol];
#pragma unroll
    for (int r = 0; r < 8; ++r) {
      int ml = mbase + kh8 + r;
      float h = siluf(a2L[r] + bL);
      float g = sigmf(a2G[r] + bG);
      size_t eid = (size_t)(e0 + ml);
      float en = edge_feat[eid * DD + ncol] + h * g;
      e_out[eid * DD + ncol] = en;          // output e_new
      xin[ml][DD + ncol] = (_Float16)en;    // splice into node-message input
    }
  }
  __syncthreads();

  // ===== node-message gated-MLP, layer 0 =====
  v8f mL = {}, mG = {};
#pragma unroll
  for (int s = 0; s < 6; ++s) {
    v16h a = load_a<DIN>(xin, mbase, lane, kh8, s * 32);
    mL = wmma_f16(a, load_b(nLW0, s, nt, lane), mL);
    mG = wmma_f16(a, load_b(nGW0, s, nt, lane), mG);
  }
  {
    float bL = nLb0[ncol], bG = nGb0[ncol];
#pragma unroll
    for (int r = 0; r < 8; ++r) {
      int ml = mbase + kh8 + r;
      hidL[ml][ncol] = (_Float16)siluf(mL[r] + bL);
      hidG[ml][ncol] = (_Float16)siluf(mG[r] + bG);
    }
  }
  __syncthreads();

  // ===== node-message gated-MLP, layer 1 + scatter-sum =====
  v8f m2L = {}, m2G = {};
#pragma unroll
  for (int s = 0; s < 2; ++s) {
    m2L = wmma_f16(load_a<DD>(hidL, mbase, lane, kh8, s * 32),
                   load_b(nLW1, s, nt, lane), m2L);
    m2G = wmma_f16(load_a<DD>(hidG, mbase, lane, kh8, s * 32),
                   load_b(nGW1, s, nt, lane), m2G);
  }
  {
    float bL = nLb1[ncol], bG = nGb1[ncol];
#pragma unroll
    for (int r = 0; r < 8; ++r) {
      int ml = mbase + kh8 + r;
      float msg = siluf(m2L[r] + bL) * sigmf(m2G[r] + bG);
      atomicAdd(&agg[(size_t)dIdx[ml] * DD + ncol], msg);
    }
  }
}

// ---------------- node finalize: h_new = node_feat + agg @ Wout ----------------
__global__ __launch_bounds__(256) void node_kernel(
    const float* __restrict__ node_feat, const float* __restrict__ agg,
    const _Float16* __restrict__ woutswz, float* __restrict__ h_out) {
  const int tid = threadIdx.x;
  const int lane = tid & 31;
  const int wave = tid >> 5;
  const int mt = wave >> 2, nt = wave & 3;
  const int n0 = blockIdx.x * 32;
  const int kh8 = (lane >> 4) << 3;
  const int ncol = nt * 16 + (lane & 15);
  int node_a = n0 + mt * 16 + (lane & 15);
  node_a = node_a < NN ? node_a : NN - 1;   // clamp (uniform-safe)

  v8f acc = {};
#pragma unroll
  for (int s = 0; s < 2; ++s) {
    FragH f;
#pragma unroll
    for (int v = 0; v < 8; ++v) {
      int kp = (v < 4) ? (kh8 + 2 * v) : (16 + kh8 + 2 * (v - 4));
      float2 xv = *(const float2*)&agg[(size_t)node_a * DD + s * 32 + kp];
      union { _Float16 h[2]; unsigned int u; } pk;
      pk.h[0] = (_Float16)xv.x; pk.h[1] = (_Float16)xv.y;
      f.u[v] = pk.u;
    }
    acc = wmma_f16(f.h, load_b(woutswz, s, nt, lane), acc);
  }
#pragma unroll
  for (int r = 0; r < 8; ++r) {
    int nr = n0 + mt * 16 + kh8 + r;
    if (nr < NN)
      h_out[(size_t)nr * DD + ncol] =
          node_feat[(size_t)nr * DD + ncol] + acc[r];
  }
}

// ---------------- launch ----------------
extern "C" void kernel_launch(void* const* d_in, const int* in_sizes, int n_in,
                              void* d_out, int out_size, void* d_ws, size_t ws_size,
                              hipStream_t stream) {
  const float* node_feat = (const float*)d_in[0];
  const float* edge_feat = (const float*)d_in[1];
  const int*   src       = (const int*)d_in[2];
  const int*   dst       = (const int*)d_in[3];
  const float* eLW0 = (const float*)d_in[4];
  const float* eLb0 = (const float*)d_in[5];
  const float* eLW1 = (const float*)d_in[6];
  const float* eLb1 = (const float*)d_in[7];
  const float* eGW0 = (const float*)d_in[8];
  const float* eGb0 = (const float*)d_in[9];
  const float* eGW1 = (const float*)d_in[10];
  const float* eGb1 = (const float*)d_in[11];
  const float* nLW0 = (const float*)d_in[12];
  const float* nLb0 = (const float*)d_in[13];
  const float* nLW1 = (const float*)d_in[14];
  const float* nLb1 = (const float*)d_in[15];
  const float* nGW0 = (const float*)d_in[16];
  const float* nGb0 = (const float*)d_in[17];
  const float* nGW1 = (const float*)d_in[18];
  const float* nGb1 = (const float*)d_in[19];
  const float* Wout = (const float*)d_in[20];

  float* h_out = (float*)d_out;                          // [30000,64]
  float* e_out = (float*)d_out + (size_t)NN * DD;        // [480000,64]

  float*    agg  = (float*)d_ws;                         // 7.68 MB
  _Float16* wswz = (_Float16*)((char*)d_ws + (size_t)NN * DD * sizeof(float));

  // zero scatter buffer
  zero_f32<<<(NN * DD + 255) / 256, 256, 0, stream>>>(agg, NN * DD);

  // swizzle all weights into WMMA-fragment order (f16); L2-resident after this
  swz_weight<<<(DIN * DD + 255) / 256, 256, 0, stream>>>(eLW0, wswz + 0,     DIN);
  swz_weight<<<(DIN * DD + 255) / 256, 256, 0, stream>>>(eGW0, wswz + 12288, DIN);
  swz_weight<<<(DD  * DD + 255) / 256, 256, 0, stream>>>(eLW1, wswz + 24576, DD);
  swz_weight<<<(DD  * DD + 255) / 256, 256, 0, stream>>>(eGW1, wswz + 28672, DD);
  swz_weight<<<(DIN * DD + 255) / 256, 256, 0, stream>>>(nLW0, wswz + 32768, DIN);
  swz_weight<<<(DIN * DD + 255) / 256, 256, 0, stream>>>(nGW0, wswz + 45056, DIN);
  swz_weight<<<(DD  * DD + 255) / 256, 256, 0, stream>>>(nLW1, wswz + 57344, DD);
  swz_weight<<<(DD  * DD + 255) / 256, 256, 0, stream>>>(nGW1, wswz + 61440, DD);
  swz_weight<<<(DD  * DD + 255) / 256, 256, 0, stream>>>(Wout, wswz + 65536, DD);

  // fused edge pipeline: 480000/32 = 15000 workgroups, 32 WMMAs per wave
  edge_kernel<<<NE / ET, 256, 0, stream>>>(
      node_feat, edge_feat, src, dst,
      eLb0, eLb1, eGb0, eGb1, nLb0, nLb1, nGb0, nGb1,
      wswz, e_out, agg);

  // node residual output
  node_kernel<<<(NN + 31) / 32, 256, 0, stream>>>(node_feat, agg, wswz + 65536,
                                                  h_out);
}
